// MultiHeadLatentAttention_9457517985809
// MI455X (gfx1250) — compile-verified, matrix operand order checked by reading, not run on hardware
//
#include <hip/hip_runtime.h>
#include <hip/hip_bf16.h>

// ---------------------------------------------------------------------------
// MLA forward for MI455X (gfx1250, wave32, WMMA bf16 16x16x32 + TDM staging)
// ---------------------------------------------------------------------------

#define B_DIM 2
#define S_LEN 2048
#define D_MODEL 2048
#define N_HEADS 16
#define QK_ROPE 64
#define QK_NOPE 128
#define KV_LORA 512
#define V_DIM 128
#define QK_DIM (QK_ROPE + QK_NOPE)            // 192
#define ROWS (B_DIM * S_LEN)                   // 4096
#define RMS_EPS 1.1920928955078125e-07f

typedef __bf16 bf16;
typedef __attribute__((ext_vector_type(16))) __bf16 v16bf;
typedef __attribute__((ext_vector_type(8)))  __bf16 v8bf;
typedef __attribute__((ext_vector_type(8)))  float  v8f;
typedef __attribute__((ext_vector_type(4)))  unsigned int v4u;
typedef __attribute__((ext_vector_type(8)))  unsigned int v8u;

// ---- helpers ---------------------------------------------------------------

__device__ __forceinline__ bf16 to_bf16(float f) {
    union { float f; unsigned u; } in; in.f = f;
    unsigned u = (in.u + 0x7FFFu + ((in.u >> 16) & 1u)) >> 16;  // RNE
    union { unsigned short s; bf16 b; } out; out.s = (unsigned short)u;
    return out.b;
}

// Load a 16x32 bf16 fragment (A layout; B layout identical with Bt[N][K]).
// lane l: row = l&15, half = l>>4; elems 0..7 = K(k0+half*8..), 8..15 = K(k0+16+half*8..)
__device__ __forceinline__ v16bf frag_ld(const bf16* base, int ld, int k0) {
    int lane = threadIdx.x & 31;
    int r = lane & 15, half = lane >> 4;
    const bf16* p = base + (size_t)r * ld + k0 + half * 8;
    v8bf lo = *(const v8bf*)p;
    v8bf hi = *(const v8bf*)(p + 16);
    v16bf f;
#pragma unroll
    for (int i = 0; i < 8; ++i) { f[i] = lo[i]; f[i + 8] = hi[i]; }
    return f;
}

__device__ __forceinline__ v8f wmma_bf16(v16bf a, v16bf b, v8f c) {
    return __builtin_amdgcn_wmma_f32_16x16x32_bf16(false, a, false, b,
                                                   (short)0, c, false, false);
}

__device__ __forceinline__ unsigned rfl(unsigned x) {
    return (unsigned)__builtin_amdgcn_readfirstlane((int)x);
}

// Tensor Data Mover: DMA a 64-row x 32-elem bf16 tile (row stride K elems)
// from global memory into LDS at byte offset lds_off. 2-D descriptor,
// D# group0/group1 packed per CDNA5 ISA 8.3/8.4.
__device__ __forceinline__ void tdm_load_b_tile(const bf16* gsrc, unsigned lds_off,
                                                int K) {
    unsigned long long ga = (unsigned long long)(uintptr_t)gsrc;
    v4u g0;
    g0.x = rfl(1u);                                   // count=1, user descriptor
    g0.y = rfl(lds_off);                              // lds_addr (bytes)
    g0.z = rfl((unsigned)(ga & 0xFFFFFFFFu));         // global_addr[31:0]
    g0.w = rfl(((unsigned)(ga >> 32) & 0x01FFFFFFu) | 0x80000000u); // ga[56:32] | type=2
    v8u g1;
    g1[0] = rfl(1u << 16);                            // data_size=1 (2 bytes)
    g1[1] = rfl(((unsigned)K & 0xFFFFu) << 16);       // tensor_dim0[15:0]
    g1[2] = rfl(((unsigned)K >> 16) & 0xFFFFu);       // tensor_dim0[31:16], tdim1 lo=0
    g1[3] = rfl((32u << 16) | 0x0010u);               // tile_dim0=32, tensor_dim1=1<<20
    g1[4] = rfl(64u);                                 // tile_dim1=64, tile_dim2=0
    g1[5] = rfl((unsigned)K);                         // tensor_dim0_stride[31:0]
    g1[6] = rfl(0u);
    g1[7] = rfl(0u);
    asm volatile("tensor_load_to_lds %0, %1" : : "s"(g0), "s"(g1) : "memory");
}

__device__ __forceinline__ void rope_pair(float x1, float x2, int s, int i,
                                          float& o1, float& o2) {
    float inv = powf(10000.0f, -((float)(2 * i)) * (1.0f / 64.0f));
    float ang = (float)s * inv;
    float c = cosf(ang), sn = sinf(ang);
    o1 = x1 * c - x2 * sn;
    o2 = x2 * c + x1 * sn;
}

// ---- conversion kernels ----------------------------------------------------

__global__ void k_f32_to_bf16(const float* __restrict__ in,
                              bf16* __restrict__ out, int n) {
    int i = blockIdx.x * 256 + threadIdx.x;
    if (i < n) out[i] = to_bf16(in[i]);
}

// W[K][N] (f32) -> Wt[N][K] (bf16)
__global__ void k_transpose_bf16(const float* __restrict__ in,
                                 bf16* __restrict__ out, int K, int N) {
    int id = blockIdx.x * 256 + threadIdx.x;
    if (id >= K * N) return;
    int k = id / N, n = id - k * N;
    out[(size_t)n * K + k] = to_bf16(in[id]);
}

// ---- generic bf16 GEMM: C[M][N] = A[M][K] * Bt[N][K]^T ---------------------
// block = 256 threads (8 waves) covering 128 rows x 64 cols; the 64x32 B tile
// is staged in LDS by the TDM (double buffered, TENSORcnt); A fragments and
// LDS B fragments are register ping-ponged so WMMAs overlap outstanding loads.

__global__ __launch_bounds__(256) void k_gemm_bf16(const bf16* __restrict__ A,
                                                   const bf16* __restrict__ Bt,
                                                   float* __restrict__ C,
                                                   int M, int N, int K) {
    __shared__ __align__(16) bf16 bstage[2][64 * 32];
    int lane = threadIdx.x & 31, w = threadIdx.x >> 5;
    int m0 = blockIdx.y * 128 + w * 16;
    int n0 = blockIdx.x * 64;
    const bf16* Abase = A + (size_t)m0 * K;
    const bf16* Bbase = Bt + (size_t)n0 * K;

    if (w == 0) {
        tdm_load_b_tile(Bbase, (unsigned)(uintptr_t)&bstage[0][0], K);
        __builtin_amdgcn_s_wait_tensorcnt(0);
    }
    __syncthreads();

    v8f acc[4];
#pragma unroll
    for (int t = 0; t < 4; ++t)
#pragma unroll
        for (int i = 0; i < 8; ++i) acc[t][i] = 0.0f;

    v16bf a_cur = frag_ld(Abase, K, 0);

    int p = 0;
    for (int k0 = 0; k0 < K; k0 += 32, p ^= 1) {
        int kn = k0 + 32;
        bool more = kn < K;
        if (more && w == 0)
            tdm_load_b_tile(Bbase + kn, (unsigned)(uintptr_t)&bstage[p ^ 1][0], K);
        v16bf a_nxt = frag_ld(Abase, K, more ? kn : 0);  // prefetch next A frag
        const bf16* bl = &bstage[p][0];
        v16bf b_cur = frag_ld(bl, 32, 0);
#pragma unroll
        for (int t = 0; t < 4; ++t) {
            v16bf b_nxt;
            if (t < 3) b_nxt = frag_ld(bl + (t + 1) * 16 * 32, 32, 0);  // B ping-pong
            acc[t] = wmma_bf16(a_cur, b_cur, acc[t]);
            b_cur = b_nxt;
        }
        a_cur = a_nxt;
        if (more && w == 0) __builtin_amdgcn_s_wait_tensorcnt(0);
        __syncthreads();
    }

    int half = lane >> 4, col = lane & 15;
#pragma unroll
    for (int t = 0; t < 4; ++t)
#pragma unroll
        for (int r = 0; r < 8; ++r)
            C[(size_t)(m0 + r + 8 * half) * N + n0 + t * 16 + col] = acc[t][r];
}

// ---- q epilogue: RoPE on dims 128..191, emit bf16 [B][H][S][192] -----------

__global__ __launch_bounds__(192) void k_rope_q(const float* __restrict__ Q,
                                                bf16* __restrict__ Qb) {
    int rowh = blockIdx.x;               // (b*S + s) * H + h
    int h = rowh & (N_HEADS - 1);
    int row = rowh >> 4;                 // b*S + s
    int b = row / S_LEN, s = row - b * S_LEN;
    int d = threadIdx.x;
    const float* qrow = Q + (size_t)row * (N_HEADS * QK_DIM) + h * QK_DIM;
    float val;
    if (d < QK_NOPE) {
        val = qrow[d];
    } else {
        int e = d - QK_NOPE, i = e >> 1;
        float o1, o2;
        rope_pair(qrow[QK_NOPE + 2 * i], qrow[QK_NOPE + 2 * i + 1], s, i, o1, o2);
        val = (e & 1) ? o2 : o1;
    }
    Qb[((size_t)(b * N_HEADS + h) * S_LEN + s) * QK_DIM + d] = to_bf16(val);
}

// ---- kv_a epilogue: RMS-norm c_kv (bf16) + RoPE k_rope (bf16) --------------

__global__ __launch_bounds__(256) void k_kva_post(const float* __restrict__ KVa,
                                                  const float* __restrict__ normw,
                                                  bf16* __restrict__ ckv,
                                                  bf16* __restrict__ kr) {
    int row = blockIdx.x;                 // b*S + s
    int s = row & (S_LEN - 1);
    int tid = threadIdx.x;
    __shared__ float red[256];
    const float* src = KVa + (size_t)row * (KV_LORA + QK_ROPE);
    float ss = 0.0f;
    for (int j = tid; j < KV_LORA; j += 256) { float c = src[j]; ss += c * c; }
    red[tid] = ss;
    __syncthreads();
    for (int off = 128; off; off >>= 1) {
        if (tid < off) red[tid] += red[tid + off];
        __syncthreads();
    }
    float rms = rsqrtf(red[0] * (1.0f / KV_LORA) + RMS_EPS);
    for (int j = tid; j < KV_LORA; j += 256)
        ckv[(size_t)row * KV_LORA + j] = to_bf16(src[j] * rms * normw[j]);
    if (tid < QK_ROPE) {
        int e = tid, i = e >> 1;
        float o1, o2;
        rope_pair(src[KV_LORA + 2 * i], src[KV_LORA + 2 * i + 1], s, i, o1, o2);
        kr[(size_t)row * QK_ROPE + e] = to_bf16((e & 1) ? o2 : o1);
    }
}

// ---- kv_b epilogue: assemble K=[k_nope|k_rope] bf16 [B][H][S][192],
//      V transposed bf16 [B][H][128][S] ------------------------------------

__global__ __launch_bounds__(320) void k_kvb_post(const float* __restrict__ KVb,
                                                  const bf16* __restrict__ kr,
                                                  bf16* __restrict__ Kb,
                                                  bf16* __restrict__ Vt) {
    int rowh = blockIdx.x;
    int h = rowh & (N_HEADS - 1);
    int row = rowh >> 4;
    int b = row / S_LEN, s = row - b * S_LEN;
    int bh = b * N_HEADS + h;
    int tid = threadIdx.x;
    const float* src = KVb + (size_t)row * (N_HEADS * (QK_NOPE + V_DIM)) + h * (QK_NOPE + V_DIM);
    if (tid < QK_NOPE) {
        Kb[((size_t)bh * S_LEN + s) * QK_DIM + tid] = to_bf16(src[tid]);
    } else if (tid < QK_DIM) {
        Kb[((size_t)bh * S_LEN + s) * QK_DIM + tid] = kr[(size_t)row * QK_ROPE + tid - QK_NOPE];
    } else {
        int d = tid - QK_DIM;
        Vt[((size_t)bh * V_DIM + d) * S_LEN + s] = to_bf16(src[QK_NOPE + d]);
    }
}

// ---- flash attention: 1 wave per (b, h, 16-query tile) ---------------------

__global__ __launch_bounds__(32) void k_attn(const bf16* __restrict__ Qb,
                                             const bf16* __restrict__ Kb,
                                             const bf16* __restrict__ Vt,
                                             bf16* __restrict__ O) {
    int q0 = blockIdx.x * 16;
    int h = blockIdx.y, b = blockIdx.z;
    int bh = b * N_HEADS + h;
    int lane = threadIdx.x;
    int half = lane >> 4, col = lane & 15;
    __shared__ __align__(16) bf16 pbuf[16 * 32];

    // Q fragments for the whole tile (16 x 192)
    v16bf qf[6];
    const bf16* qbase = Qb + ((size_t)bh * S_LEN + q0) * QK_DIM;
#pragma unroll
    for (int d = 0; d < 6; ++d) qf[d] = frag_ld(qbase, QK_DIM, d * 32);

    v8f acc[8];
#pragma unroll
    for (int t = 0; t < 8; ++t)
#pragma unroll
        for (int i = 0; i < 8; ++i) acc[t][i] = 0.0f;
    float m_[8], l_[8];
#pragma unroll
    for (int r = 0; r < 8; ++r) { m_[r] = -3.0e38f; l_[r] = 0.0f; }

    const float scale = 0.07216878364870322f;  // 1/sqrt(192)
    int nchunk = ((q0 + 15) >> 5) + 1;          // causal: skip fully-masked chunks

    const bf16* Kbh = Kb + (size_t)bh * S_LEN * QK_DIM;
    const bf16* Vbh = Vt + (size_t)bh * V_DIM * S_LEN;

    // prefetch first K fragments
    v16bf kb0 = frag_ld(Kbh, QK_DIM, 0);
    v16bf kb1 = frag_ld(Kbh + 16 * QK_DIM, QK_DIM, 0);

    for (int ch = 0; ch < nchunk; ++ch) {
        int kbase = ch * 32;
        int knb = (ch + 1 < nchunk) ? kbase + 32 : kbase;  // next-chunk prefetch base
        const bf16* k0p = Kbh + (size_t)kbase * QK_DIM;
        const bf16* k1p = k0p + 16 * QK_DIM;
        v8f s0, s1;
#pragma unroll
        for (int i = 0; i < 8; ++i) { s0[i] = 0.0f; s1[i] = 0.0f; }
#pragma unroll
        for (int d = 0; d < 6; ++d) {
            v16bf nb0, nb1;
            if (d < 5) {
                nb0 = frag_ld(k0p, QK_DIM, (d + 1) * 32);
                nb1 = frag_ld(k1p, QK_DIM, (d + 1) * 32);
            } else {
                nb0 = frag_ld(Kbh + (size_t)knb * QK_DIM, QK_DIM, 0);
                nb1 = frag_ld(Kbh + (size_t)(knb + 16) * QK_DIM, QK_DIM, 0);
            }
            s0 = wmma_bf16(qf[d], kb0, s0);
            s1 = wmma_bf16(qf[d], kb1, s1);
            kb0 = nb0; kb1 = nb1;
        }
        // issue first V fragment early so it hides under the softmax math
        v16bf vb = frag_ld(Vbh, S_LEN, kbase);
        // online softmax per owned row
#pragma unroll
        for (int r = 0; r < 8; ++r) {
            int qg = q0 + r + 8 * half;
            float v0 = (kbase + col <= qg) ? s0[r] * scale : -3.0e38f;
            float v1 = (kbase + 16 + col <= qg) ? s1[r] * scale : -3.0e38f;
            float mx = fmaxf(v0, v1);
#pragma unroll
            for (int off = 8; off; off >>= 1) mx = fmaxf(mx, __shfl_xor(mx, off, 32));
            float mnew = fmaxf(m_[r], mx);
            float corr = __expf(m_[r] - mnew);
            float p0 = __expf(v0 - mnew);
            float p1 = __expf(v1 - mnew);
            float rs = p0 + p1;
#pragma unroll
            for (int off = 8; off; off >>= 1) rs += __shfl_xor(rs, off, 32);
            l_[r] = l_[r] * corr + rs;
            m_[r] = mnew;
#pragma unroll
            for (int t = 0; t < 8; ++t) acc[t][r] *= corr;
            pbuf[(r + 8 * half) * 32 + col] = to_bf16(p0);
            pbuf[(r + 8 * half) * 32 + 16 + col] = to_bf16(p1);
        }
        __syncthreads();
        v16bf pa = frag_ld(pbuf, 32, 0);
#pragma unroll
        for (int t = 0; t < 8; ++t) {
            v16bf vn;
            if (t < 7) vn = frag_ld(Vbh + (size_t)(t + 1) * 16 * S_LEN, S_LEN, kbase);
            acc[t] = wmma_bf16(pa, vb, acc[t]);
            vb = vn;
        }
        __syncthreads();
    }
    // epilogue: normalize, write bf16 [b][s][h*128 + d]
#pragma unroll
    for (int r = 0; r < 8; ++r) {
        float inv = 1.0f / l_[r];
        int sg = q0 + r + 8 * half;
#pragma unroll
        for (int t = 0; t < 8; ++t)
            O[((size_t)b * S_LEN + sg) * (N_HEADS * V_DIM) + h * V_DIM + t * 16 + col] =
                to_bf16(acc[t][r] * inv);
    }
}

// ---------------------------------------------------------------------------

static inline size_t align256(size_t x) { return (x + 255) & ~(size_t)255; }

extern "C" void kernel_launch(void* const* d_in, const int* in_sizes, int n_in,
                              void* d_out, int out_size, void* d_ws, size_t ws_size,
                              hipStream_t stream) {
    const float* x      = (const float*)d_in[0];
    const float* Wq     = (const float*)d_in[1];
    const float* Wkv_a  = (const float*)d_in[2];
    const float* kvnw   = (const float*)d_in[3];
    const float* Wkv_b  = (const float*)d_in[4];
    const float* Wo     = (const float*)d_in[5];
    float* out = (float*)d_out;

    char* ws = (char*)d_ws;
    size_t off = 0;
    auto carve = [&](size_t bytes) { char* p = ws + off; off += align256(bytes); return p; };

    bf16*  xb    = (bf16*)carve((size_t)ROWS * D_MODEL * 2);
    bf16*  WqT   = (bf16*)carve((size_t)(N_HEADS * QK_DIM) * D_MODEL * 2);
    bf16*  WkvaT = (bf16*)carve((size_t)(KV_LORA + QK_ROPE) * D_MODEL * 2);
    bf16*  WkvbT = (bf16*)carve((size_t)(N_HEADS * (QK_NOPE + V_DIM)) * KV_LORA * 2);
    bf16*  WoT   = (bf16*)carve((size_t)D_MODEL * (N_HEADS * V_DIM) * 2);
    // big f32 scratch shared by Q (48MB) then KVb (64MB)
    float* big   = (float*)carve((size_t)ROWS * (N_HEADS * (QK_NOPE + V_DIM)) * 4);
    float* KVa   = (float*)carve((size_t)ROWS * (KV_LORA + QK_ROPE) * 4);
    bf16*  ckv   = (bf16*)carve((size_t)ROWS * KV_LORA * 2);
    bf16*  kr    = (bf16*)carve((size_t)ROWS * QK_ROPE * 2);
    bf16*  Qb    = (bf16*)carve((size_t)B_DIM * N_HEADS * S_LEN * QK_DIM * 2);
    bf16*  Kb    = (bf16*)carve((size_t)B_DIM * N_HEADS * S_LEN * QK_DIM * 2);
    bf16*  Vt    = (bf16*)carve((size_t)B_DIM * N_HEADS * V_DIM * S_LEN * 2);
    bf16*  Ob    = (bf16*)carve((size_t)ROWS * (N_HEADS * V_DIM) * 2);

    // 1) conversions
    {
        int n = ROWS * D_MODEL;
        k_f32_to_bf16<<<(n + 255) / 256, 256, 0, stream>>>(x, xb, n);
    }
    {
        int n = D_MODEL * (N_HEADS * QK_DIM);
        k_transpose_bf16<<<(n + 255) / 256, 256, 0, stream>>>(Wq, WqT, D_MODEL, N_HEADS * QK_DIM);
    }
    {
        int n = D_MODEL * (KV_LORA + QK_ROPE);
        k_transpose_bf16<<<(n + 255) / 256, 256, 0, stream>>>(Wkv_a, WkvaT, D_MODEL, KV_LORA + QK_ROPE);
    }
    {
        int n = KV_LORA * (N_HEADS * (QK_NOPE + V_DIM));
        k_transpose_bf16<<<(n + 255) / 256, 256, 0, stream>>>(Wkv_b, WkvbT, KV_LORA, N_HEADS * (QK_NOPE + V_DIM));
    }
    {
        int n = (N_HEADS * V_DIM) * D_MODEL;
        k_transpose_bf16<<<(n + 255) / 256, 256, 0, stream>>>(Wo, WoT, N_HEADS * V_DIM, D_MODEL);
    }

    // 2) kv_a = x @ Wkv_a  (M=4096, N=576, K=2048)
    k_gemm_bf16<<<dim3((KV_LORA + QK_ROPE) / 64, ROWS / 128), 256, 0, stream>>>(
        xb, WkvaT, KVa, ROWS, KV_LORA + QK_ROPE, D_MODEL);

    // 3) q = x @ Wq  (M=4096, N=3072, K=2048) -> big scratch
    k_gemm_bf16<<<dim3((N_HEADS * QK_DIM) / 64, ROWS / 128), 256, 0, stream>>>(
        xb, WqT, big, ROWS, N_HEADS * QK_DIM, D_MODEL);

    // 4) rope(q) -> Qb [B][H][S][192]
    k_rope_q<<<ROWS * N_HEADS, 192, 0, stream>>>(big, Qb);

    // 5) RMS-norm + rope(k_rope)
    k_kva_post<<<ROWS, 256, 0, stream>>>(KVa, kvnw, ckv, kr);

    // 6) kv = ckv @ Wkv_b  (M=4096, N=4096, K=512) -> big scratch (Q now dead)
    k_gemm_bf16<<<dim3((N_HEADS * (QK_NOPE + V_DIM)) / 64, ROWS / 128), 256, 0, stream>>>(
        ckv, WkvbT, big, ROWS, N_HEADS * (QK_NOPE + V_DIM), KV_LORA);

    // 7) assemble K, V^T
    k_kvb_post<<<ROWS * N_HEADS, 320, 0, stream>>>(big, kr, Kb, Vt);

    // 8) flash attention -> Ob bf16 [B][S][2048]
    k_attn<<<dim3(S_LEN / 16, N_HEADS, B_DIM), 32, 0, stream>>>(Qb, Kb, Vt, Ob);

    // 9) out = Ob @ Wo  (M=4096, N=2048, K=2048)
    k_gemm_bf16<<<dim3(D_MODEL / 64, ROWS / 128), 256, 0, stream>>>(
        Ob, WoT, out, ROWS, D_MODEL, N_HEADS * V_DIM);
}
